// ConvolutionStack_37297495999124
// MI455X (gfx1250) — matrix operand order repaired and checked by reading, not compiled, not run on hardware
//
#include <hip/hip_runtime.h>
#include <hip/hip_bf16.h>
#include <math.h>

// ---------------------------------------------------------------------------
// WaveNet-ish dilated conv stack on gfx1250 (MI455X), bf16x3 split WMMA.
// Round 2: cooperative LDS staging of conv B-fragments (8x less global x
// traffic + 8x less cvt VALU), hardware tanh/rcp for the gated activation.
// ---------------------------------------------------------------------------

typedef __attribute__((ext_vector_type(16))) __bf16 v16bf;
typedef __attribute__((ext_vector_type(8)))  float  v8f;

#define Cc   256
#define C2   512
#define Tt   8192
#define Bb   4
#define Ll   10
#define NCOL (Bb * Tt)          // 32768 GEMM columns
#define BCT  (Bb * Cc * Tt)     // 8,388,608 floats per [B,C,T] tensor

// Pre-swizzled weight fragment sizes (bf16 elements)
#define WA_PER_LAYER (3 * 32 * 8 * 32 * 16)   // taps * mtiles * ksteps * lanes * 16
#define WO_PER_LAYER (16 * 8 * 32 * 16)

#if __has_builtin(__builtin_amdgcn_tanhf)
#define FAST_TANH(x) __builtin_amdgcn_tanhf(x)
#else
#define FAST_TANH(x) tanhf(x)
#endif

static __device__ inline v8f zero8() {
    v8f z;
    #pragma unroll
    for (int i = 0; i < 8; i++) z[i] = 0.0f;
    return z;
}

// ---------------------------------------------------------------------------
// Weight prep: fp32 -> (hi, lo) bf16, pre-swizzled into WMMA 16x32 A-fragment
// lane layout (contiguous K, this file's own order; B staging matches): lane = 16*half + m%16, element j -> K:
//   k = 16*half + j
// ---------------------------------------------------------------------------
__global__ void prep_conv_weights(const float* __restrict__ wconv,
                                  __bf16* __restrict__ wAh,
                                  __bf16* __restrict__ wAl) {
    int idx = blockIdx.x * 256 + threadIdx.x;
    const int TOT = Ll * WA_PER_LAYER;
    if (idx >= TOT) return;
    int j    = idx & 15;
    int lane = (idx >> 4) & 31;
    int ks   = (idx >> 9) & 7;
    int mt   = (idx >> 12) & 31;
    int rest = idx >> 17;          // i*3 + kk
    int kk   = rest % 3;
    int i    = rest / 3;
    int m    = mt * 16 + (lane & 15);
    int half = lane >> 4;
    int k    = 16 * half + j;
    int c    = ks * 32 + k;
    float v  = wconv[((size_t)(i * C2 + m) * Cc + c) * 3 + kk];
    __bf16 hi = (__bf16)v;
    wAh[idx] = hi;
    wAl[idx] = (__bf16)(v - (float)hi);
}

__global__ void prep_out_weights(const float* __restrict__ wout,
                                 __bf16* __restrict__ wOh,
                                 __bf16* __restrict__ wOl) {
    int idx = blockIdx.x * 256 + threadIdx.x;
    const int TOT = Ll * WO_PER_LAYER;
    if (idx >= TOT) return;
    int j    = idx & 15;
    int lane = (idx >> 4) & 31;
    int ks   = (idx >> 9) & 7;
    int mt   = (idx >> 12) & 15;
    int i    = idx >> 16;
    int m    = mt * 16 + (lane & 15);
    int half = lane >> 4;
    int k    = 16 * half + j;
    int c    = ks * 32 + k;
    float v  = wout[(size_t)(i * Cc + m) * Cc + c];
    __bf16 hi = (__bf16)v;
    wOh[idx] = hi;
    wOl[idx] = (__bf16)(v - (float)hi);
}

// ---------------------------------------------------------------------------
// Fused layer kernel: dilated conv (3-tap GEMM) -> gated act -> 1x1 -> residual
// Block = 256 threads = 8 waves. Block owns 32 timesteps x all 512 z rows.
// ---------------------------------------------------------------------------
__global__ __launch_bounds__(256)
void wavenet_layer(const float* __restrict__ xin, float* __restrict__ xout,
                   float* __restrict__ skip,
                   const __bf16* __restrict__ wAh, const __bf16* __restrict__ wAl,
                   const __bf16* __restrict__ wOh, const __bf16* __restrict__ wOl,
                   const float* __restrict__ bconv, const float* __restrict__ bout,
                   int dil, int last) {
    // 64 KB LDS: (a) conv-phase bf16 B staging (32 KB), (b) z f32 [512][32],
    // (c) h bf16 hi/lo staging for the 1x1. Phases separated by barriers.
    __shared__ float smem[C2 * 32];

    const int tid  = threadIdx.x;
    const int lane = tid & 31;
    const int w    = tid >> 5;                 // wave id 0..7
    const int half = lane >> 4;
    const int lcol = lane & 15;
    const int n0   = blockIdx.x * 32;          // global column
    const int b    = n0 >> 13;                 // / 8192
    const int t0   = n0 & (Tt - 1);

    // ---------------- conv GEMM: z[512 x 32] = sum_tap W_tap @ x_shift ------
    v8f acc[4][2];
    #pragma unroll
    for (int mt = 0; mt < 4; mt++)
        #pragma unroll
        for (int ct = 0; ct < 2; ct++) acc[mt][ct] = zero8();

    const int mbase = w * 64;                  // wave owns z rows [mbase, mbase+64)
    __bf16* stage = (__bf16*)smem;             // hi at [0,8192), lo at [8192,16384)

    for (int kk = 0; kk < 3; kk++) {
        const int tsh = (2 - kk) * dil;        // causal tap shift
        const int tb  = t0 - tsh;              // block-uniform base timestep
        const __bf16* wAhk = wAh + (size_t)kk * (32 * 8 * 32 * 16);
        const __bf16* wAlk = wAl + (size_t)kk * (32 * 8 * 32 * 16);

        __syncthreads();   // previous tap's fragment reads complete
        // Cooperative staging: thread owns channel c = ks*32 + 16*half + j and
        // streams 32 consecutive timesteps; writes pre-swizzled bf16 hi/lo.
        {
            int sks  = tid >> 5;
            int shlf = (tid >> 4) & 1;
            int sj   = tid & 15;
            int c    = sks * 32 + 16 * shlf + sj;
            const float* xrow = xin + (size_t)(b * Cc + c) * Tt + tb;
            if (tb >= 0) {  // common path: uniform, branch-free contiguous loads
                #pragma unroll
                for (int q = 0; q < 32; q++) {
                    float v = xrow[q];
                    __bf16 h = (__bf16)v;
                    int idx = ((sks * 2 + (q >> 4)) * 32 + ((q & 15) + 16 * shlf)) * 16 + sj;
                    stage[idx]        = h;
                    stage[8192 + idx] = (__bf16)(v - (float)h);
                }
            } else {
                #pragma unroll
                for (int q = 0; q < 32; q++) {
                    float v = (tb + q >= 0) ? xrow[q] : 0.0f;
                    __bf16 h = (__bf16)v;
                    int idx = ((sks * 2 + (q >> 4)) * 32 + ((q & 15) + 16 * shlf)) * 16 + sj;
                    stage[idx]        = h;
                    stage[8192 + idx] = (__bf16)(v - (float)h);
                }
            }
        }
        __syncthreads();   // staged B visible to all waves

        #pragma unroll
        for (int ks = 0; ks < 8; ks++) {
            v16bf bH[2], bL[2];
            #pragma unroll
            for (int ct = 0; ct < 2; ct++) {
                int off = ((ks * 2 + ct) * 32 + lane) * 16;
                bH[ct] = *(const v16bf*)(stage + off);
                bL[ct] = *(const v16bf*)(stage + 8192 + off);
            }
            #pragma unroll
            for (int mt = 0; mt < 4; mt++) {
                size_t aoff = ((size_t)((mbase >> 4) + mt) * 8 + ks) * 32 * 16
                              + (size_t)lane * 16;
                v16bf aH = *(const v16bf*)(wAhk + aoff);
                v16bf aL = *(const v16bf*)(wAlk + aoff);
                #pragma unroll
                for (int ct = 0; ct < 2; ct++) {
                    acc[mt][ct] = __builtin_amdgcn_wmma_f32_16x16x32_bf16(
                        false, aH, false, bH[ct], (short)0, acc[mt][ct], false, false);
                    acc[mt][ct] = __builtin_amdgcn_wmma_f32_16x16x32_bf16(
                        false, aH, false, bL[ct], (short)0, acc[mt][ct], false, false);
                    acc[mt][ct] = __builtin_amdgcn_wmma_f32_16x16x32_bf16(
                        false, aL, false, bH[ct], (short)0, acc[mt][ct], false, false);
                }
            }
        }
    }
    __syncthreads();       // last tap's fragment reads complete before z lands

    // ---------------- z + bias -> LDS (C/D layout: row = r + 8*half) --------
    #pragma unroll
    for (int mt = 0; mt < 4; mt++) {
        #pragma unroll
        for (int r = 0; r < 8; r++) {
            int m = mbase + mt * 16 + r + 8 * half;
            float bb = bconv[m];
            #pragma unroll
            for (int ct = 0; ct < 2; ct++)
                smem[m * 32 + ct * 16 + lcol] = acc[mt][ct][r] + bb;
        }
    }
    __syncthreads();

    // ---------------- gated activation + skip write -------------------------
    float hv[32];
    {
        const int c = tid;
        #pragma unroll
        for (int n = 0; n < 32; n++) {
            float zt = smem[c * 32 + n];
            float zs = smem[(c + 256) * 32 + n];
            hv[n] = FAST_TANH(zt) * __builtin_amdgcn_rcpf(1.0f + __expf(-zs));
        }
        float4* sk = (float4*)(skip + (size_t)(b * Cc + c) * Tt + t0);
        #pragma unroll
        for (int q = 0; q < 8; q++)
            sk[q] = make_float4(hv[4*q], hv[4*q+1], hv[4*q+2], hv[4*q+3]);
    }

    if (last) {  // final layer: out = h, x_next = h + x (uniform exit)
        const int c = tid;
        const float* xi = xin  + (size_t)(b * Cc + c) * Tt + t0;
        float*       xo = xout + (size_t)(b * Cc + c) * Tt + t0;
        #pragma unroll
        for (int n = 0; n < 32; n++) xo[n] = hv[n] + xi[n];
        return;
    }

    __syncthreads();
    // ---------------- repack h as pre-swizzled bf16 hi/lo B-fragments -------
    {
        __bf16* hb = (__bf16*)smem;   // hi at [0,8192), lo at [8192,16384)
        const int c  = tid;
        const int ksi = c >> 5, hlf = (c >> 4) & 1, j = c & 15;
        #pragma unroll
        for (int n = 0; n < 32; n++) {
            int idx = ((ksi * 2 + (n >> 4)) * 32 + ((n & 15) + 16 * hlf)) * 16 + j;
            __bf16 hh = (__bf16)hv[n];
            hb[idx]        = hh;
            hb[8192 + idx] = (__bf16)(hv[n] - (float)hh);
        }
    }
    __syncthreads();

    // ---------------- 1x1 GEMM + bias + residual ----------------------------
    {
        const __bf16* hb = (const __bf16*)smem;
        v8f oacc[2][2];
        #pragma unroll
        for (int mt = 0; mt < 2; mt++)
            #pragma unroll
            for (int ct = 0; ct < 2; ct++) oacc[mt][ct] = zero8();
        const int m1 = w * 32;                 // wave owns out rows [m1, m1+32)
        #pragma unroll
        for (int ks = 0; ks < 8; ks++) {
            v16bf bH[2], bL[2];
            #pragma unroll
            for (int ct = 0; ct < 2; ct++) {
                int off = ((ks * 2 + ct) * 32 + lane) * 16;
                bH[ct] = *(const v16bf*)(hb + off);
                bL[ct] = *(const v16bf*)(hb + 8192 + off);
            }
            #pragma unroll
            for (int mt = 0; mt < 2; mt++) {
                size_t aoff = ((size_t)((m1 >> 4) + mt) * 8 + ks) * 32 * 16
                              + (size_t)lane * 16;
                v16bf aH = *(const v16bf*)(wOh + aoff);
                v16bf aL = *(const v16bf*)(wOl + aoff);
                #pragma unroll
                for (int ct = 0; ct < 2; ct++) {
                    oacc[mt][ct] = __builtin_amdgcn_wmma_f32_16x16x32_bf16(
                        false, aH, false, bH[ct], (short)0, oacc[mt][ct], false, false);
                    oacc[mt][ct] = __builtin_amdgcn_wmma_f32_16x16x32_bf16(
                        false, aH, false, bL[ct], (short)0, oacc[mt][ct], false, false);
                    oacc[mt][ct] = __builtin_amdgcn_wmma_f32_16x16x32_bf16(
                        false, aL, false, bH[ct], (short)0, oacc[mt][ct], false, false);
                }
            }
        }
        #pragma unroll
        for (int mt = 0; mt < 2; mt++) {
            #pragma unroll
            for (int r = 0; r < 8; r++) {
                int m = m1 + mt * 16 + r + 8 * half;
                float bb = bout[m];
                #pragma unroll
                for (int ct = 0; ct < 2; ct++) {
                    int t = t0 + ct * 16 + lcol;
                    size_t gi = (size_t)(b * Cc + m) * Tt + t;
                    xout[gi] = oacc[mt][ct][r] + bb + xin[gi];
                }
            }
        }
    }
}

// ---------------------------------------------------------------------------
// Host launcher
// ---------------------------------------------------------------------------
extern "C" void kernel_launch(void* const* d_in, const int* in_sizes, int n_in,
                              void* d_out, int out_size, void* d_ws, size_t ws_size,
                              hipStream_t stream) {
    (void)in_sizes; (void)n_in; (void)out_size; (void)ws_size;
    static const int DIL[Ll] = {1, 2, 4, 8, 16, 32, 64, 128, 256, 512};

    const float* input = (const float*)d_in[0];   // [B,C,T]
    const float* wconv = (const float*)d_in[1];   // [L,2C,C,K]
    const float* bconv = (const float*)d_in[2];   // [L,2C]
    const float* wout  = (const float*)d_in[3];   // [L,C,C]
    const float* bout  = (const float*)d_in[4];   // [L,C]

    float* out_x = (float*)d_out;                 // [B,C,T]
    float* skips = out_x + (size_t)BCT;           // [L,B,C,T]

    // Workspace layout: x ping buffer, then pre-swizzled bf16 weights.
    char* ws = (char*)d_ws;
    float*  x_ws = (float*)ws;                                   // 33,554,432 B
    __bf16* wAh  = (__bf16*)(ws + (size_t)BCT * 4);
    __bf16* wAl  = wAh + (size_t)Ll * WA_PER_LAYER;
    __bf16* wOh  = wAl + (size_t)Ll * WA_PER_LAYER;
    __bf16* wOl  = wOh + (size_t)Ll * WO_PER_LAYER;

    {
        int totA = Ll * WA_PER_LAYER;
        int totO = Ll * WO_PER_LAYER;
        prep_conv_weights<<<(totA + 255) / 256, 256, 0, stream>>>(wconv, wAh, wAl);
        prep_out_weights<<<(totO + 255) / 256, 256, 0, stream>>>(wout, wOh, wOl);
    }

    const int nblocks = NCOL / 32;                // 1024 blocks per layer
    for (int i = 0; i < Ll; i++) {
        const float* xin  = (i == 0) ? input : ((i & 1) ? x_ws : out_x);
        float*       xout = (i & 1) ? out_x : x_ws;   // layer 9 (odd) -> out_x
        wavenet_layer<<<nblocks, 256, 0, stream>>>(
            xin, xout, skips + (size_t)i * BCT,
            wAh + (size_t)i * WA_PER_LAYER, wAl + (size_t)i * WA_PER_LAYER,
            wOh + (size_t)i * WO_PER_LAYER, wOl + (size_t)i * WO_PER_LAYER,
            bconv + (size_t)i * C2, bout + (size_t)i * Cc,
            DIL[i], (i == Ll - 1) ? 1 : 0);
    }
}
